// MPNN_18880676233594
// MI455X (gfx1250) — matrix operand order repaired and checked by reading, not compiled, not run on hardware
//
#include <hip/hip_runtime.h>
#include <hip/hip_bf16.h>
#include <math.h>

// ---------------- problem constants ----------------
#define NNODES 20000
#define NEDGES 40000
#define HID    64
#define FATOM  62
#define FBOND  6
#define NGRAPH 512

typedef __attribute__((ext_vector_type(16))) __bf16 v16bf;
typedef __attribute__((ext_vector_type(8)))  float  v8f;

static __device__ __forceinline__ float selu_f(float x) {
  const float alpha = 1.6732632423543772f;
  const float scale = 1.0507009873554805f;
  return x > 0.f ? scale * x : scale * alpha * (__expf(x) - 1.f);
}

// A-fragment K index for 16-bit A 16x32 (contiguous, matches B loads): lane-group g (0:lanes0-15,
// 1:lanes16-31), element t (0..15) -> K = 16*g + t in [0,32).
static __device__ __forceinline__ int a_kmap(int g, int t) {
  return 16 * g + t;
}

static __device__ __forceinline__ v8f vzero8() {
  v8f z = {0.f, 0.f, 0.f, 0.f, 0.f, 0.f, 0.f, 0.f};
  return z;
}

// ---------------- prep kernels ----------------
__global__ void zero_kernel(float* __restrict__ p, int n) {
  int i = blockIdx.x * blockDim.x + threadIdx.x;
  if (i < n) p[i] = 0.f;
}

__global__ void init_h_kernel(const float* __restrict__ nf,
                              float* __restrict__ h, __bf16* __restrict__ h_bf) {
  int idx = blockIdx.x * blockDim.x + threadIdx.x;
  if (idx >= NNODES * HID) return;
  int n = idx >> 6, i = idx & 63;
  float v = (i < FATOM) ? nf[n * FATOM + i] : 0.f;
  h[idx] = v;
  h_bf[idx] = (__bf16)v;
}

__global__ void edge_mlp_kernel(const float* __restrict__ ef, const float* __restrict__ Wm,
                                const float* __restrict__ bm, __bf16* __restrict__ mlp_bf) {
  int idx = blockIdx.x * blockDim.x + threadIdx.x;
  if (idx >= NEDGES * HID) return;
  int e = idx >> 6, i = idx & 63;
  float acc = bm[i];
#pragma unroll
  for (int f = 0; f < FBOND; ++f) acc += ef[e * FBOND + f] * Wm[f * HID + i];
  mlp_bf[idx] = (__bf16)fmaxf(acc, 0.f);
}

// EwT[i][k*64+j] = Ew[k][i*64+j]   (Ew is [64, 4096] row-major)
__global__ void ewt_kernel(const float* __restrict__ Ew, __bf16* __restrict__ EwT) {
  int idx = blockIdx.x * blockDim.x + threadIdx.x;
  if (idx >= HID * HID * HID) return;
  int i = idx >> 12;
  int f = idx & 4095;
  int k = f >> 6, j = f & 63;
  EwT[idx] = (__bf16)Ew[k * 4096 + i * 64 + j];
}

// WT[i][k] = W[k][i]  (64x64), fp32 -> bf16
__global__ void wtrans_kernel(const float* __restrict__ W, __bf16* __restrict__ WT) {
  int idx = blockIdx.x * blockDim.x + threadIdx.x;
  if (idx >= HID * HID) return;
  int i = idx >> 6, k = idx & 63;
  WT[idx] = (__bf16)W[k * HID + i];
}

// ---------------- fused edge-message GEMM + scatter ----------------
// message[dom[e],:] += (mlp[e] (x) hj[e]) @ EwR, A rows built in registers.
// Each wave: 32 edges = two 16-row A tiles sharing B fragments (halves L2
// traffic on EwT). B is ping-pong double-buffered across a 2x-unrolled K loop
// so L2 latency overlaps WMMA with zero register-rotation moves.
#define EG_WAVES 2
#define EG_ROWS  32
__global__ void __launch_bounds__(32 * EG_WAVES)
edge_gemm_kernel(const __bf16* __restrict__ mlp_bf, const __bf16* __restrict__ h_bf,
                 const int* __restrict__ edge_range, const int* __restrict__ edge_domain,
                 const __bf16* __restrict__ EwT, float* __restrict__ msg) {
  __shared__ __align__(16) __bf16 s_h[EG_WAVES][EG_ROWS][HID];
  __shared__ __align__(16) __bf16 s_m[EG_WAVES][EG_ROWS][HID];
  __shared__ int s_dom[EG_WAVES][EG_ROWS];
  const int wave = threadIdx.x >> 5;
  const int lane = threadIdx.x & 31;
  const int tile = blockIdx.x * EG_WAVES + wave;   // grid sized exactly
  const int e0 = tile * EG_ROWS;
  const int m = lane & 15;        // A row within tile / B column within n-tile
  const int g = lane >> 4;        // lane group

  // Stage gathered h rows, mlp rows, and domains for 32 edges (lane L -> edge e0+L)
  {
    const int e = e0 + lane;
    const int r = edge_range[e];
    s_dom[wave][lane] = edge_domain[e];
    const float4* srch = (const float4*)(h_bf + (size_t)r * HID);
    const float4* srcm = (const float4*)(mlp_bf + (size_t)e * HID);
    float4* dsth = (float4*)(&s_h[wave][lane][0]);
    float4* dstm = (float4*)(&s_m[wave][lane][0]);
#pragma unroll
    for (int q = 0; q < 8; ++q) {
      dsth[q] = srch[q];
      dstm[q] = srcm[q];
    }
  }
  __syncthreads();

  v8f acc0[4], acc1[4];
#pragma unroll
  for (int nt = 0; nt < 4; ++nt) {
    acc0[nt] = vzero8();
    acc1[nt] = vzero8();
  }

  // B fragment base for this lane: col = m, K-halves at g*16, contiguous in EwT
  const __bf16* bbase = EwT + (size_t)m * 4096 + g * 16;
  v16bf bufA[4], bufB[4];
#pragma unroll
  for (int nt = 0; nt < 4; ++nt)
    bufA[nt] = *(const v16bf*)(bbase + (size_t)nt * 16 * 4096);

  // one (k,j)-chunk of the outer-product GEMM: build two A frags, 8 WMMAs
  auto do_chunk = [&](int c, const v16bf* bb) {
    const int kc = c >> 1;          // fixed k for this 32-wide (k,j) chunk
    const int j0 = (c & 1) << 5;    // j base (0 or 32)
    const float mv0 = (float)s_m[wave][m][kc];
    const float mv1 = (float)s_m[wave][16 + m][kc];
    v16bf a0, a1;
#pragma unroll
    for (int t = 0; t < 16; ++t) {
      const int K = a_kmap(g, t);
      const float h0 = (float)s_h[wave][m][j0 + K];
      const float h1 = (float)s_h[wave][16 + m][j0 + K];
      a0[t] = (__bf16)(mv0 * h0);
      a1[t] = (__bf16)(mv1 * h1);
    }
#pragma unroll
    for (int nt = 0; nt < 4; ++nt) {
      acc0[nt] = __builtin_amdgcn_wmma_f32_16x16x32_bf16(
          false, a0, false, bb[nt], (short)0, acc0[nt], false, false);
      acc1[nt] = __builtin_amdgcn_wmma_f32_16x16x32_bf16(
          false, a1, false, bb[nt], (short)0, acc1[nt], false, false);
    }
  };

#pragma unroll 1
  for (int c = 0; c < 128; c += 2) {
    // load chunk c+1 into bufB, compute chunk c from bufA
#pragma unroll
    for (int nt = 0; nt < 4; ++nt)
      bufB[nt] = *(const v16bf*)(bbase + (size_t)nt * 16 * 4096 + (c + 1) * 32);
    do_chunk(c, bufA);
    // load chunk c+2 into bufA (last iteration wraps to 0; harmless), compute c+1
    const int c2 = (c + 2) & 127;
#pragma unroll
    for (int nt = 0; nt < 4; ++nt)
      bufA[nt] = *(const v16bf*)(bbase + (size_t)nt * 16 * 4096 + c2 * 32);
    do_chunk(c + 1, bufB);
  }

  // fused segment_sum: scatter-add rows into message[dom[e],:]
#pragma unroll
  for (int r = 0; r < 8; ++r) {
    const int mo = r + 8 * g;       // D layout: VGPR r -> row r / r+8 by lane group
    const int d0 = s_dom[wave][mo];
    const int d1 = s_dom[wave][16 + mo];
#pragma unroll
    for (int nt = 0; nt < 4; ++nt) {
      const int n = nt * 16 + m;
      atomicAdd(&msg[(size_t)d0 * HID + n], acc0[nt][r]);
      atomicAdd(&msg[(size_t)d1 * HID + n], acc1[nt][r]);
    }
  }
}

// ---------------- node update: h = selu(msg @ Wu + bu + h) ----------------
__global__ void __launch_bounds__(32)
node_update_kernel(const float* __restrict__ msg, const __bf16* __restrict__ WuT,
                   const float* __restrict__ bu, float* __restrict__ h,
                   __bf16* __restrict__ h_bf) {
  const int n0 = blockIdx.x * 16;
  const int lane = threadIdx.x & 31;
  const int m = lane & 15, g = lane >> 4;
  v8f acc[4];
#pragma unroll
  for (int nt = 0; nt < 4; ++nt) acc[nt] = vzero8();

#pragma unroll
  for (int kc = 0; kc < 2; ++kc) {
    const int k0 = kc * 32;
    v16bf a;
#pragma unroll
    for (int t = 0; t < 16; ++t)
      a[t] = (__bf16)msg[(size_t)(n0 + m) * HID + k0 + a_kmap(g, t)];
#pragma unroll
    for (int nt = 0; nt < 4; ++nt) {
      const v16bf b = *(const v16bf*)(WuT + (nt * 16 + m) * HID + k0 + g * 16);
      acc[nt] = __builtin_amdgcn_wmma_f32_16x16x32_bf16(
          false, a, false, b, (short)0, acc[nt], false, false);
    }
  }
#pragma unroll
  for (int nt = 0; nt < 4; ++nt)
#pragma unroll
    for (int r = 0; r < 8; ++r) {
      const int mo = r + 8 * g, n = nt * 16 + m;
      const size_t o = (size_t)(n0 + mo) * HID + n;
      const float x = selu_f(acc[nt][r] + bu[n] + h[o]);
      h[o] = x;
      h_bf[o] = (__bf16)x;
    }
}

// ---------------- readout: selu((h@Wae+bae)@WR+bR) -> segment_sum ----------------
__global__ void __launch_bounds__(32)
readout_kernel(const __bf16* __restrict__ h_bf, const __bf16* __restrict__ WaeT,
               const float* __restrict__ bae, const __bf16* __restrict__ WRT,
               const float* __restrict__ bR, const int* __restrict__ graph_id,
               float* __restrict__ gsum) {
  __shared__ __align__(16) __bf16 s_ae[16][HID];
  const int n0 = blockIdx.x * 16;
  const int lane = threadIdx.x & 31;
  const int m = lane & 15, g = lane >> 4;

  v8f acc[4];
#pragma unroll
  for (int nt = 0; nt < 4; ++nt) acc[nt] = vzero8();

#pragma unroll
  for (int kc = 0; kc < 2; ++kc) {
    const int k0 = kc * 32;
    v16bf a;
#pragma unroll
    for (int t = 0; t < 16; ++t)
      a[t] = h_bf[(size_t)(n0 + m) * HID + k0 + a_kmap(g, t)];
#pragma unroll
    for (int nt = 0; nt < 4; ++nt) {
      const v16bf b = *(const v16bf*)(WaeT + (nt * 16 + m) * HID + k0 + g * 16);
      acc[nt] = __builtin_amdgcn_wmma_f32_16x16x32_bf16(
          false, a, false, b, (short)0, acc[nt], false, false);
    }
  }
  // atom_embedding -> LDS (re-fragment D layout into A layout)
#pragma unroll
  for (int nt = 0; nt < 4; ++nt)
#pragma unroll
    for (int r = 0; r < 8; ++r)
      s_ae[r + 8 * g][nt * 16 + m] = (__bf16)(acc[nt][r] + bae[nt * 16 + m]);
  __syncthreads();

#pragma unroll
  for (int nt = 0; nt < 4; ++nt) acc[nt] = vzero8();
#pragma unroll
  for (int kc = 0; kc < 2; ++kc) {
    const int k0 = kc * 32;
    v16bf a;
#pragma unroll
    for (int t = 0; t < 16; ++t) a[t] = s_ae[m][k0 + a_kmap(g, t)];
#pragma unroll
    for (int nt = 0; nt < 4; ++nt) {
      const v16bf b = *(const v16bf*)(WRT + (nt * 16 + m) * HID + k0 + g * 16);
      acc[nt] = __builtin_amdgcn_wmma_f32_16x16x32_bf16(
          false, a, false, b, (short)0, acc[nt], false, false);
    }
  }
#pragma unroll
  for (int nt = 0; nt < 4; ++nt)
#pragma unroll
    for (int r = 0; r < 8; ++r) {
      const int mo = r + 8 * g, n = nt * 16 + m;
      const float x = selu_f(acc[nt][r] + bR[n]);
      atomicAdd(&gsum[(size_t)graph_id[n0 + mo] * HID + n], x);
    }
}

// ---------------- final head: tanh -> relu GEMV -> dot ----------------
__global__ void __launch_bounds__(64)
head_kernel(const float* __restrict__ gsum, const float* __restrict__ Wmlp,
            const float* __restrict__ bmlp, const float* __restrict__ Wout,
            const float* __restrict__ bout, float* __restrict__ out) {
  __shared__ float ge[HID];
  __shared__ float red[HID];
  const int gq = blockIdx.x, t = threadIdx.x;
  ge[t] = tanhf(gsum[gq * HID + t]);
  __syncthreads();
  float acc = bmlp[t];
#pragma unroll 8
  for (int k = 0; k < HID; ++k) acc += ge[k] * Wmlp[k * HID + t];
  red[t] = fmaxf(acc, 0.f) * Wout[t];
  __syncthreads();
  for (int s = 32; s > 0; s >>= 1) {
    if (t < s) red[t] += red[t + s];
    __syncthreads();
  }
  if (t == 0) out[gq] = red[0] + bout[0];
}

// ---------------- host side ----------------
extern "C" void kernel_launch(void* const* d_in, const int* in_sizes, int n_in,
                              void* d_out, int out_size, void* d_ws, size_t ws_size,
                              hipStream_t stream) {
  const float* node_features = (const float*)d_in[0];
  const float* edge_features = (const float*)d_in[1];
  const int*   edge_domain   = (const int*)d_in[2];
  const int*   edge_range    = (const int*)d_in[3];
  const int*   graph_id      = (const int*)d_in[4];
  const float* Wm   = (const float*)d_in[5];
  const float* bm   = (const float*)d_in[6];
  const float* Ew   = (const float*)d_in[7];
  const float* Wu0  = (const float*)d_in[8];
  const float* bu0  = (const float*)d_in[9];
  const float* Wu1  = (const float*)d_in[10];
  const float* bu1  = (const float*)d_in[11];
  const float* Wae  = (const float*)d_in[12];
  const float* bae  = (const float*)d_in[13];
  const float* WR   = (const float*)d_in[14];
  const float* bR   = (const float*)d_in[15];
  const float* Wmlp = (const float*)d_in[16];
  const float* bmlp = (const float*)d_in[17];
  const float* Wout = (const float*)d_in[18];
  const float* bout = (const float*)d_in[19];
  float* out = (float*)d_out;

  char* ws = (char*)d_ws;
  float*  h      = (float*)(ws + 0);              // 5,120,000 B
  __bf16* h_bf   = (__bf16*)(ws + 5120000);       // 2,560,000 B
  __bf16* mlp_bf = (__bf16*)(ws + 7680000);       // 5,120,000 B
  __bf16* EwT    = (__bf16*)(ws + 12800000);      //   524,288 B
  __bf16* WuT0   = (__bf16*)(ws + 13324288);      // 4 x 8,192 B
  __bf16* WuT1   = WuT0 + 4096;
  __bf16* WaeT   = WuT0 + 8192;
  __bf16* WRT    = WuT0 + 12288;
  float*  msg    = (float*)(ws + 13357056);       // 5,120,000 B
  float*  gsum   = (float*)(ws + 18477056);       //   131,072 B

  // prep
  init_h_kernel<<<(NNODES * HID + 255) / 256, 256, 0, stream>>>(node_features, h, h_bf);
  edge_mlp_kernel<<<(NEDGES * HID + 255) / 256, 256, 0, stream>>>(edge_features, Wm, bm, mlp_bf);
  ewt_kernel<<<(HID * HID * HID + 255) / 256, 256, 0, stream>>>(Ew, EwT);
  wtrans_kernel<<<16, 256, 0, stream>>>(Wu0, WuT0);
  wtrans_kernel<<<16, 256, 0, stream>>>(Wu1, WuT1);
  wtrans_kernel<<<16, 256, 0, stream>>>(Wae, WaeT);
  wtrans_kernel<<<16, 256, 0, stream>>>(WR, WRT);

  // T = 2 message-passing steps
  const __bf16* WuT[2] = {WuT0, WuT1};
  const float*  bu[2]  = {bu0, bu1};
  for (int t = 0; t < 2; ++t) {
    zero_kernel<<<(NNODES * HID + 255) / 256, 256, 0, stream>>>(msg, NNODES * HID);
    edge_gemm_kernel<<<NEDGES / (EG_ROWS * EG_WAVES), 32 * EG_WAVES, 0, stream>>>(
        mlp_bf, h_bf, edge_range, edge_domain, EwT, msg);
    node_update_kernel<<<NNODES / 16, 32, 0, stream>>>(msg, WuT[t], bu[t], h, h_bf);
  }

  // readout + head
  zero_kernel<<<(NGRAPH * HID + 255) / 256, 256, 0, stream>>>(gsum, NGRAPH * HID);
  readout_kernel<<<NNODES / 16, 32, 0, stream>>>(h_bf, WaeT, bae, WRT, bR, graph_id, gsum);
  head_kernel<<<NGRAPH, 64, 0, stream>>>(gsum, Wmlp, bmlp, Wout, bout, out);
}
